// Spatial_Temporal_Transformer_47785806135951
// MI455X (gfx1250) — matrix operand order repaired and checked by reading, not compiled, not run on hardware
//
#include <hip/hip_runtime.h>

typedef __attribute__((ext_vector_type(16))) __bf16 v16bf;
typedef __attribute__((ext_vector_type(8)))  float  v8f;
typedef __attribute__((ext_vector_type(4)))  unsigned v4u;
typedef __attribute__((ext_vector_type(8)))  int    v8i;
typedef __attribute__((ext_vector_type(4)))  int    v4i;

#define DD   256
#define PP   16
#define CEH  32
#define HPH  4
#define HSH  4
#define NPP  16
#define CCH  128
#define SEQ  512
#define BSZ  32
#define TOK  (BSZ*SEQ)     // 16384

// ---------- helpers ----------
__device__ __forceinline__ __bf16 f2bf(float f) {
  unsigned u = __builtin_bit_cast(unsigned, f);
  unsigned r = u + 0x7FFFu + ((u >> 16) & 1u);
  unsigned short h = (unsigned short)(r >> 16);
  return __builtin_bit_cast(__bf16, h);
}
__device__ __forceinline__ int amap(int e, int half) {
  // 16-bit A 16x32 layout (ISA 7.12.2)
  return (e < 8) ? (half * 8 + e) : (16 + half * 8 + (e - 8));
}
__device__ __forceinline__ int bmap(int e, int half) {
  // 16-bit B 32x16 layout (same k order as A)
  return (e < 8) ? (half * 8 + e) : (16 + half * 8 + (e - 8));
}
__device__ __forceinline__ v8f wmma_bf16(v16bf a, v16bf b, v8f c) {
  return __builtin_amdgcn_wmma_f32_16x16x32_bf16(false, a, false, b, (short)0, c,
                                                 false, false);
}
__device__ __forceinline__ unsigned lds_off(const void* p) {
  // low 32 bits of flat shared address == DS offset
  return (unsigned)(size_t)p;
}
__device__ __forceinline__ unsigned rfl(unsigned x) {
  return (unsigned)__builtin_amdgcn_readfirstlane((int)x);
}

// TDM: DMA a 2D bf16 tile [tileY rows x tileX elems], row stride strideElems,
// from global memory into LDS at byte offset ldsOff. (CDNA5 ISA D# layout)
__device__ __forceinline__ void tdm_load_2d_bf16(unsigned ldsOff, const void* gptr,
                                                 unsigned tileX, unsigned tileY,
                                                 unsigned strideElems) {
  size_t ga = (size_t)gptr;
  unsigned galo = (unsigned)(ga & 0xFFFFFFFFu);
  unsigned gahi = (unsigned)((ga >> 32) & 0x01FFFFFFu) | (2u << 30); // type=2
  v4u g0;
  g0[0] = rfl(1u);                       // count=1, user mode
  g0[1] = rfl(ldsOff);                   // lds_addr (bytes)
  g0[2] = rfl(galo);                     // global_addr[31:0]
  g0[3] = rfl(gahi);                     // global_addr[56:32] | type
  v8i g1;
  g1[0] = (int)rfl(1u << 16);            // data_size=1 (2 bytes), no multicast
  g1[1] = (int)rfl((tileX & 0xFFFFu) << 16);                 // tensor_dim0[15:0]
  g1[2] = (int)rfl((tileX >> 16) | ((tileY & 0xFFFFu) << 16)); // td0 hi | td1 lo
  g1[3] = (int)rfl((tileY >> 16) | ((tileX & 0xFFFFu) << 16)); // td1 hi | tile_dim0
  g1[4] = (int)rfl(tileY & 0xFFFFu);     // tile_dim1 (tile_dim2 = 0)
  g1[5] = (int)rfl(strideElems);         // tensor_dim0_stride[31:0]
  g1[6] = 0;                             // stride hi | tensor_dim1_stride lo
  g1[7] = 0;
  v4i z4;
  z4[0] = 0; z4[1] = 0; z4[2] = 0; z4[3] = 0;
#if defined(__clang_major__) && (__clang_major__ >= 23)
  v8i z8;
  z8[0] = 0; z8[1] = 0; z8[2] = 0; z8[3] = 0;
  z8[4] = 0; z8[5] = 0; z8[6] = 0; z8[7] = 0;
  __builtin_amdgcn_tensor_load_to_lds(g0, g1, z4, z4, z8, 0);
#else
  __builtin_amdgcn_tensor_load_to_lds(g0, g1, z4, z4, 0);
#endif
}

// ---------- 1) f32 -> bf16 convert ----------
__global__ void k_cvt(const float* __restrict__ in, __bf16* __restrict__ out, int n) {
  int i = blockIdx.x * blockDim.x + threadIdx.x;
  if (i < n) out[i] = f2bf(in[i]);
}

// ---------- 2) generic WMMA GEMM: out[M,N] = A[M,K](bf16) @ W[N,K]^T + bias ----------
// A-tile staged via Tensor Data Mover; B-tile staged manually (f32->bf16 convert).
// Requires M,K multiples of 64/32 for the TDM tile (true for all launches here).
template <bool OUTF32>
__global__ void k_gemm(const __bf16* __restrict__ A, int lda,
                       const float* __restrict__ W, int ldw,
                       const float* __restrict__ bias,
                       void* __restrict__ out, int ldo,
                       int M, int N, int K) {
  __shared__ __bf16 As[64][32];
  __shared__ __bf16 Bs[64][32];
  const int t = threadIdx.x;
  const int wave = t >> 5, lane = t & 31, half = lane >> 4, ln = lane & 15;
  const int m0 = blockIdx.y * 64, n0 = blockIdx.x * 64;
  const __bf16 BZ = f2bf(0.0f);
  const unsigned asOff = lds_off(&As[0][0]);

  v8f acc[4];
#pragma unroll
  for (int nt = 0; nt < 4; ++nt)
#pragma unroll
    for (int r = 0; r < 8; ++r) acc[nt][r] = 0.0f;

  const int lrow = t >> 1;            // 0..63
  const int kb = (t & 1) * 16;        // 0 or 16

  for (int k0 = 0; k0 < K; k0 += 32) {
    // async A tile: 64 rows x 32 bf16, row stride lda
    if (t == 0)
      tdm_load_2d_bf16(asOff, &A[(size_t)m0 * lda + k0], 32u, 64u, (unsigned)lda);
    if (k0 + 32 < K)
      __builtin_prefetch(&W[(size_t)(n0 + lrow) * ldw + k0 + 32 + kb], 0, 3);
#pragma unroll
    for (int i = 0; i < 16; ++i) {
      int n = n0 + lrow, k = k0 + kb + i;
      Bs[lrow][kb + i] = (n < N && k < K) ? f2bf(W[(size_t)n * ldw + k]) : BZ;
    }
    __builtin_amdgcn_s_wait_tensorcnt(0);
    asm volatile("" ::: "memory");
    __syncthreads();
    v16bf a;
#pragma unroll
    for (int e = 0; e < 16; ++e) a[e] = As[wave * 16 + ln][amap(e, half)];
#pragma unroll
    for (int nt = 0; nt < 4; ++nt) {
      v16bf b;
#pragma unroll
      for (int e = 0; e < 16; ++e) b[e] = Bs[nt * 16 + ln][bmap(e, half)];
      acc[nt] = wmma_bf16(a, b, acc[nt]);
    }
    __syncthreads();
  }
#pragma unroll
  for (int nt = 0; nt < 4; ++nt)
#pragma unroll
    for (int r = 0; r < 8; ++r) {
      int m = m0 + wave * 16 + r + 8 * half;
      int n = n0 + nt * 16 + ln;
      if (m < M && n < N) {
        float v = acc[nt][r] + (bias ? bias[n] : 0.0f);
        if constexpr (OUTF32)
          ((float*)out)[(size_t)m * ldo + n] = v;
        else
          ((__bf16*)out)[(size_t)m * ldo + n] = f2bf(v);
      }
    }
}

// ---------- 3) attention: one WG per (b,h,32-row tile), 2 waves ----------
__global__ void k_attn(const __bf16* __restrict__ Q, const __bf16* __restrict__ Km,
                       const __bf16* __restrict__ Vm, __bf16* __restrict__ O) {
  extern __shared__ char smem[];
  float*  Sc = (float*)smem;                       // [32][512] scores
  __bf16* Pl = (__bf16*)(smem + 32 * 512 * 4);     // [32][512] probs bf16

  const int b = blockIdx.z, h = blockIdx.y, rt = blockIdx.x;
  const int ld = HSH * DD;                         // 1024
  const size_t base = (size_t)b * SEQ * ld;
  const int hoff = h * DD;
  const int wave = threadIdx.x >> 5, lane = threadIdx.x & 31;
  const int half = lane >> 4, ln = lane & 15;
  const int row0 = rt * 32 + wave * 16;            // sequence row of this wave's tile

  // preload Q fragments (K = 256 -> 8 steps)
  v16bf qa[8];
#pragma unroll
  for (int ks = 0; ks < 8; ++ks)
#pragma unroll
    for (int e = 0; e < 16; ++e)
      qa[ks][e] = Q[base + (size_t)(row0 + ln) * ld + hoff + ks * 32 + amap(e, half)];

  const float inv = 1.0f / 256.0f;                 // reference divides by D
  for (int nt = 0; nt < 32; ++nt) {                // 512 key cols
    v8f c;
#pragma unroll
    for (int r = 0; r < 8; ++r) c[r] = 0.0f;
#pragma unroll
    for (int ks = 0; ks < 8; ++ks) {
      v16bf kb;
#pragma unroll
      for (int e = 0; e < 16; ++e)
        kb[e] = Km[base + (size_t)(nt * 16 + ln) * ld + hoff + ks * 32 + bmap(e, half)];
      c = wmma_bf16(qa[ks], kb, c);
    }
#pragma unroll
    for (int r = 0; r < 8; ++r)
      Sc[(wave * 16 + r + 8 * half) * 512 + nt * 16 + ln] = c[r] * inv;
  }
  __syncthreads();

  // softmax: lane handles row (wave*16+ln), columns [half*256, half*256+256)
  {
    const int lr = wave * 16 + ln;
    float mx = -3.4e38f;
    for (int cidx = 0; cidx < 256; ++cidx)
      mx = fmaxf(mx, Sc[lr * 512 + half * 256 + cidx]);
    mx = fmaxf(mx, __shfl_xor(mx, 16));
    float sm = 0.0f;
    for (int cidx = 0; cidx < 256; ++cidx) {
      float e_ = __expf(Sc[lr * 512 + half * 256 + cidx] - mx);
      Sc[lr * 512 + half * 256 + cidx] = e_;
      sm += e_;
    }
    sm += __shfl_xor(sm, 16);
    float rs = 1.0f / sm;
    for (int cidx = 0; cidx < 256; ++cidx)
      Pl[lr * 512 + half * 256 + cidx] = f2bf(Sc[lr * 512 + half * 256 + cidx] * rs);
  }
  __syncthreads();

  // P @ V : 16 N-tiles over d, 16 K-steps over t
  v8f oacc[16];
#pragma unroll
  for (int nt = 0; nt < 16; ++nt)
#pragma unroll
    for (int r = 0; r < 8; ++r) oacc[nt][r] = 0.0f;

  for (int kt = 0; kt < 16; ++kt) {
    v16bf pa;
#pragma unroll
    for (int e = 0; e < 16; ++e)
      pa[e] = Pl[(wave * 16 + ln) * 512 + kt * 32 + amap(e, half)];
#pragma unroll
    for (int nt = 0; nt < 16; ++nt) {
      v16bf vb;
#pragma unroll
      for (int e = 0; e < 16; ++e)
        vb[e] = Vm[base + (size_t)(kt * 32 + bmap(e, half)) * ld + hoff + nt * 16 + ln];
      oacc[nt] = wmma_bf16(pa, vb, oacc[nt]);
    }
  }
#pragma unroll
  for (int nt = 0; nt < 16; ++nt)
#pragma unroll
    for (int r = 0; r < 8; ++r)
      O[base + (size_t)(row0 + r + 8 * half) * ld + hoff + nt * 16 + ln] =
          f2bf(oacc[nt][r]);
}

// ---------- 4) LayerNorm (mean/std ddof=1, no eps) with residual ----------
__global__ void k_ln(const float* __restrict__ a, const float* __restrict__ res,
                     float* __restrict__ outf, __bf16* __restrict__ outb, int rows) {
  const int wave = threadIdx.x >> 5, lane = threadIdx.x & 31;
  const int row = blockIdx.x * 8 + wave;
  if (row >= rows) return;
  float v[8], s = 0.0f;
#pragma unroll
  for (int i = 0; i < 8; ++i) {
    int c = i * 32 + lane;
    v[i] = a[(size_t)row * 256 + c] + res[(size_t)row * 256 + c];
    s += v[i];
  }
#pragma unroll
  for (int m = 16; m >= 1; m >>= 1) s += __shfl_xor(s, m);
  float mean = s * (1.0f / 256.0f);
  float q = 0.0f;
#pragma unroll
  for (int i = 0; i < 8; ++i) { float d = v[i] - mean; q += d * d; }
#pragma unroll
  for (int m = 16; m >= 1; m >>= 1) q += __shfl_xor(q, m);
  float rstd = rsqrtf(q * (1.0f / 255.0f));
#pragma unroll
  for (int i = 0; i < 8; ++i) {
    int c = i * 32 + lane;
    float o = (v[i] - mean) * rstd;
    outf[(size_t)row * 256 + c] = o;
    if (outb) outb[(size_t)row * 256 + c] = f2bf(o);
  }
}

// ---------- 5) fused per-token patch attention (1 wave = 1 token) ----------
__global__ void k_patch(const __bf16* __restrict__ x1b,
                        const float* __restrict__ Wqc, const float* __restrict__ bqc,
                        const float* __restrict__ Wkc, const float* __restrict__ bkc,
                        const float* __restrict__ Wvl, const float* __restrict__ bvl,
                        const float* __restrict__ Wlo, const float* __restrict__ blo,
                        __bf16* __restrict__ oc2) {
  __shared__ __bf16 QcL[4][16][32];
  __shared__ __bf16 KcL[4][16][32];
  __shared__ __bf16 VcL[4][32][32];
  __shared__ __bf16 PL[4][16][32];
  __shared__ __bf16 OcL[4][16][128];

  const int w = threadIdx.x >> 5, lane = threadIdx.x & 31;
  const int half = lane >> 4, ln = lane & 15;
  const int token = blockIdx.x * 4 + w;
  const __bf16 BZ = f2bf(0.0f);

  // zero pad regions: VcL rows 16..31, PL cols 16..31
  for (int i = lane; i < 512; i += 32) VcL[w][16 + i / 32][i % 32] = BZ;
  for (int i = lane; i < 256; i += 32) PL[w][i / 16][16 + i % 16] = BZ;
  __syncthreads();

  // A fragment of xp[16 patches][16] (K padded to 32)
  v16bf axp;
#pragma unroll
  for (int e = 0; e < 16; ++e) {
    int k = amap(e, half);
    axp[e] = (k < 16) ? x1b[(size_t)token * 256 + ln * 16 + k] : BZ;
  }

  const float* Ws[3]  = {Wqc, Wkc, Wvl};
  const float* Bsp[3] = {bqc, bkc, bvl};

  for (int h = 0; h < HPH; ++h) {
    // projections Qc/Kc/Vc for this head (2 N-tiles of 16 channels)
    for (int p = 0; p < 3; ++p) {
      __bf16(*dst)[32] = (p == 0) ? QcL[w] : (p == 1) ? KcL[w] : VcL[w];
#pragma unroll
      for (int nt = 0; nt < 2; ++nt) {
        int ch = h * 32 + nt * 16 + ln;
        v16bf wb;
#pragma unroll
        for (int e = 0; e < 16; ++e) {
          int k = bmap(e, half);
          wb[e] = (k < 16) ? f2bf(Ws[p][(size_t)ch * 16 + k]) : BZ;
        }
        v8f c;
#pragma unroll
        for (int r = 0; r < 8; ++r) c[r] = 0.0f;
        c = wmma_bf16(axp, wb, c);
        float bia = Bsp[p][ch];
#pragma unroll
        for (int r = 0; r < 8; ++r)
          dst[r + 8 * half][nt * 16 + ln] = f2bf(c[r] + bia);
      }
    }
    __syncthreads();

    // scores = Qc_h (16x32) @ Kc_h^T (32x16), scale 1/sqrt(256)
    v16bf qa, kb;
#pragma unroll
    for (int e = 0; e < 16; ++e) qa[e] = QcL[w][ln][amap(e, half)];
#pragma unroll
    for (int e = 0; e < 16; ++e) kb[e] = KcL[w][ln][bmap(e, half)];
    v8f sc;
#pragma unroll
    for (int r = 0; r < 8; ++r) sc[r] = 0.0f;
    sc = wmma_bf16(qa, kb, sc);

    // softmax over 16 key columns (16-lane reduction inside each half)
    float pv[8];
#pragma unroll
    for (int r = 0; r < 8; ++r) pv[r] = sc[r] * 0.0625f;
#pragma unroll
    for (int r = 0; r < 8; ++r) {
      float mx = pv[r];
#pragma unroll
      for (int m = 8; m >= 1; m >>= 1) mx = fmaxf(mx, __shfl_xor(mx, m));
      float e_ = __expf(pv[r] - mx);
      float sm = e_;
#pragma unroll
      for (int m = 8; m >= 1; m >>= 1) sm += __shfl_xor(sm, m);
      pv[r] = e_ / sm;
    }
#pragma unroll
    for (int r = 0; r < 8; ++r) PL[w][r + 8 * half][ln] = f2bf(pv[r]);
    __syncthreads();

    // P (16x16 pad 32) @ Vc_h (16x32 pad 32x32) -> oc_h [16x32]
    v16bf pa;
#pragma unroll
    for (int e = 0; e < 16; ++e) pa[e] = PL[w][ln][amap(e, half)];
#pragma unroll
    for (int nt = 0; nt < 2; ++nt) {
      v16bf vb;
#pragma unroll
      for (int e = 0; e < 16; ++e) vb[e] = VcL[w][bmap(e, half)][nt * 16 + ln];
      v8f c;
#pragma unroll
      for (int r = 0; r < 8; ++r) c[r] = 0.0f;
      c = wmma_bf16(pa, vb, c);
#pragma unroll
      for (int r = 0; r < 8; ++r)
        OcL[w][r + 8 * half][h * 32 + nt * 16 + ln] = f2bf(c[r]);
    }
    __syncthreads();
  }

  // oc [16x128] @ Wlo^T [128x16] + blo -> [16x16] -> flatten to oc2 row
  v8f acc;
#pragma unroll
  for (int r = 0; r < 8; ++r) acc[r] = 0.0f;
#pragma unroll
  for (int ks = 0; ks < 4; ++ks) {
    v16bf oa, lb;
#pragma unroll
    for (int e = 0; e < 16; ++e) oa[e] = OcL[w][ln][ks * 32 + amap(e, half)];
#pragma unroll
    for (int e = 0; e < 16; ++e)
      lb[e] = f2bf(Wlo[(size_t)ln * 128 + ks * 32 + bmap(e, half)]);
    acc = wmma_bf16(oa, lb, acc);
  }
  float bia = blo[ln];
#pragma unroll
  for (int r = 0; r < 8; ++r)
    oc2[(size_t)token * 256 + (r + 8 * half) * 16 + ln] = f2bf(acc[r] + bia);
}

// ---------- launch ----------
extern "C" void kernel_launch(void* const* d_in, const int* in_sizes, int n_in,
                              void* d_out, int out_size, void* d_ws, size_t ws_size,
                              hipStream_t stream) {
  const float* x     = (const float*)d_in[0];
  const float* Wq    = (const float*)d_in[1];
  const float* bq    = (const float*)d_in[2];
  const float* Wk    = (const float*)d_in[3];
  const float* bk    = (const float*)d_in[4];
  const float* Wv    = (const float*)d_in[5];
  const float* bv    = (const float*)d_in[6];
  const float* Wlin1 = (const float*)d_in[7];
  const float* blin1 = (const float*)d_in[8];
  const float* Wqc   = (const float*)d_in[9];
  const float* bqc   = (const float*)d_in[10];
  const float* Wkc   = (const float*)d_in[11];
  const float* bkc   = (const float*)d_in[12];
  const float* Wvl   = (const float*)d_in[13];
  const float* bvl   = (const float*)d_in[14];
  const float* Wlo   = (const float*)d_in[15];
  const float* blo   = (const float*)d_in[16];
  const float* Wlout = (const float*)d_in[17];
  const float* blout = (const float*)d_in[18];

  char* ws = (char*)d_ws;
  size_t off = 0;
  auto alloc = [&](size_t bytes) {
    void* p = ws + off;
    off = (off + bytes + 255) & ~(size_t)255;
    return p;
  };
  __bf16* Xb  = (__bf16*)alloc((size_t)TOK * 256 * 2);
  __bf16* Qb  = (__bf16*)alloc((size_t)TOK * 1024 * 2);
  __bf16* Kb  = (__bf16*)alloc((size_t)TOK * 1024 * 2);
  __bf16* Vb  = (__bf16*)alloc((size_t)TOK * 1024 * 2);
  __bf16* Ob  = (__bf16*)alloc((size_t)TOK * 1024 * 2);
  float*  tmp = (float*)alloc((size_t)TOK * 256 * 4);
  float*  x1f = (float*)alloc((size_t)TOK * 256 * 4);
  __bf16* x1b = (__bf16*)alloc((size_t)TOK * 256 * 2);
  __bf16* oc2 = (__bf16*)alloc((size_t)TOK * 256 * 2);

  // 1) x -> bf16
  k_cvt<<<(TOK * 256) / 256, 256, 0, stream>>>(x, Xb, TOK * 256);

  // 2) Q/K/V projections: [16384,256] x [256,1024]
  dim3 gQKV(1024 / 64, TOK / 64);
  k_gemm<false><<<gQKV, 128, 0, stream>>>(Xb, 256, Wq, 256, bq, Qb, 1024, TOK, 1024, 256);
  k_gemm<false><<<gQKV, 128, 0, stream>>>(Xb, 256, Wk, 256, bk, Kb, 1024, TOK, 1024, 256);
  k_gemm<false><<<gQKV, 128, 0, stream>>>(Xb, 256, Wv, 256, bv, Vb, 1024, TOK, 1024, 256);

  // 3) attention per (b,h,32-row tile); 96KB dynamic LDS (scores f32 + probs bf16)
  k_attn<<<dim3(SEQ / 32, HSH, BSZ), 64, 32 * 512 * 4 + 32 * 512 * 2, stream>>>(
      Qb, Kb, Vb, Ob);

  // 4) x1 = LN(O @ Wlin1^T + blin1 + x)
  k_gemm<true><<<dim3(256 / 64, TOK / 64), 128, 0, stream>>>(
      Ob, 1024, Wlin1, 1024, blin1, tmp, 256, TOK, 256, 1024);
  k_ln<<<TOK / 8, 256, 0, stream>>>(tmp, x, x1f, x1b, TOK);

  // 5) fused patch attention -> oc2 (bf16 [16384,256])
  k_patch<<<TOK / 4, 128, 0, stream>>>(x1b, Wqc, bqc, Wkc, bkc, Wvl, bvl, Wlo, blo, oc2);

  // 6) out = LN(oc2 @ Wlout^T + blout + x1)
  k_gemm<true><<<dim3(256 / 64, TOK / 64), 128, 0, stream>>>(
      oc2, 256, Wlout, 256, blout, tmp, 256, TOK, 256, 256);
  k_ln<<<TOK / 8, 256, 0, stream>>>(tmp, x1f, (float*)d_out, nullptr, TOK);
}
